// AttentionBlock_83674552860700
// MI455X (gfx1250) — compile-verified, matrix operand order checked by reading, not run on hardware
//
#include <hip/hip_runtime.h>
#include <stdint.h>

// ---------------- problem constants (fixed by reference) ----------------
#define BATCH   16
#define CDIM    512
#define TDIM    1024
#define NHEADS  8
#define CH      64          // CDIM / NHEADS
#define NGROUPS 32
#define CPG     16          // CDIM / NGROUPS

typedef __bf16 bf16;
typedef __attribute__((ext_vector_type(16))) __bf16 v16bf;
typedef __attribute__((ext_vector_type(8)))  float  v8f;
typedef __attribute__((ext_vector_type(4)))  unsigned int u32x4;
typedef int v4i __attribute__((vector_size(4 * sizeof(int))));

// ---------------- async global->LDS (CDNA5) with sync fallback ----------------
#if __has_builtin(__builtin_amdgcn_global_load_async_to_lds_b128)
#define HAVE_ASYNC 1
#else
#define HAVE_ASYNC 0
#endif

__device__ __forceinline__ void cp_b128(const bf16* gsrc, bf16* ldst) {
#if HAVE_ASYNC
  __builtin_amdgcn_global_load_async_to_lds_b128(
      (__attribute__((address_space(1))) v4i*)gsrc,
      (__attribute__((address_space(3))) v4i*)ldst, 0, 0);
#else
  *(u32x4*)ldst = *(const u32x4*)gsrc;
#endif
}

__device__ __forceinline__ void wait_async0() {
#if HAVE_ASYNC
#if __has_builtin(__builtin_amdgcn_s_wait_asynccnt)
  __builtin_amdgcn_s_wait_asynccnt(0);
#else
  asm volatile("s_wait_asynccnt 0" ::: "memory");
#endif
#endif
}

// ---------------- WMMA helper ----------------
__device__ __forceinline__ v8f wmma_bf16(v16bf a, v16bf b, v8f c) {
  return __builtin_amdgcn_wmma_f32_16x16x32_bf16(
      /*neg_a=*/false, a, /*neg_b=*/false, b,
      /*c_mod=*/(short)0, c, /*reuse_a=*/false, /*reuse_b=*/false);
}

// Fragment gather with K contiguous (pitch multiple of 8 elems keeps the two
// 16B runs aligned).  A: elem(m,k)=base[m*pitch+k]; B: elem(k,n)=base[n*pitch+k]
// Lane (half,idx): elems 0..7 = K 8*half..+7, elems 8..15 = K 16+8*half..+7.
__device__ __forceinline__ v16bf ld_frag_k(const bf16* base, int pitch) {
  const int lane = threadIdx.x & 31;
  const int half = lane >> 4;
  const int idx  = lane & 15;
  const bf16* p = base + idx * pitch + 8 * half;
  union { u32x4 u[2]; v16bf v; } t;
  t.u[0] = *(const u32x4*)(p);
  t.u[1] = *(const u32x4*)(p + 16);
  return t.v;
}

// ---------------- fp32 -> bf16 convert ----------------
__global__ __launch_bounds__(256) void f2bf_kernel(const float* __restrict__ src,
                                                   bf16* __restrict__ dst, int n) {
  int i = blockIdx.x * 256 + threadIdx.x;
  if (i < n) dst[i] = (bf16)src[i];
}

// ---------------- GroupNorm -> hT[b][t][c] (bf16) ----------------
__global__ __launch_bounds__(256) void groupnorm_kernel(const float* __restrict__ x,
                                                        const float* __restrict__ gscale,
                                                        const float* __restrict__ gbias,
                                                        bf16* __restrict__ hT) {
  const int bi = blockIdx.x / NGROUPS;
  const int g  = blockIdx.x % NGROUPS;
  const float* xp = x + ((size_t)bi * CDIM + (size_t)g * CPG) * TDIM;
  const int N = CPG * TDIM;  // 16384

  __shared__ float rs[256];
  __shared__ float rss[256];
  __shared__ bf16 buf[CPG][TDIM + 8];

  float s = 0.f, ss = 0.f;
  for (int i = threadIdx.x; i < N; i += 256) {
    float v = xp[i];
    s += v; ss += v * v;
  }
  rs[threadIdx.x] = s; rss[threadIdx.x] = ss;
  __syncthreads();
#pragma unroll
  for (int off = 128; off > 0; off >>= 1) {
    if (threadIdx.x < off) {
      rs[threadIdx.x]  += rs[threadIdx.x + off];
      rss[threadIdx.x] += rss[threadIdx.x + off];
    }
    __syncthreads();
  }
  const float mean = rs[0] * (1.f / N);
  const float var  = rss[0] * (1.f / N) - mean * mean;
  const float rstd = rsqrtf(var + 1e-5f);

  for (int i = threadIdx.x; i < N; i += 256) {
    const int cl = i >> 10;
    const int tt = i & 1023;
    const int c  = g * CPG + cl;
    float v = (xp[i] - mean) * rstd * gscale[c] + gbias[c];
    buf[cl][tt] = (bf16)v;
  }
  __syncthreads();
  for (int i = threadIdx.x; i < N; i += 256) {
    const int cl = i & 15;
    const int tt = i >> 4;
    hT[((size_t)bi * TDIM + tt) * CDIM + g * CPG + cl] = buf[cl][tt];
  }
}

// ---------------- Unified WMMA GEMM, async double-buffered ----------------
// D[m,n] = sum_k Arow(m)[k] * Brow(n)[k].  Tile 128(M) x 64(N), K-step 64.
// MODE 0 (QK):   A=hT rows (m=t), B=w_qkv rows o=(n>>7)*192+(n&127); out qkT bf16
// MODE 1 (V):    A=w_qkv rows o=(m>>6)*192+128+(m&63), B=hT rows (n=t); out v[b][c][t]
// MODE 2 (PROJ): A=w_proj rows m, B=rT rows (n=t); out fp32 + bias[m] + resid
template <int MODE>
__global__ __launch_bounds__(256) void gemm_kernel(const bf16* __restrict__ Ag,
                                                   const bf16* __restrict__ Bg,
                                                   const float* __restrict__ bias,
                                                   const float* __restrict__ resid,
                                                   void* __restrict__ outp) {
  constexpr int BM = 128, BN = 64, BK = 64, K = CDIM, NK = K / BK;
  constexpr int LP = BK + 8;  // 72 elems -> 144 B pitch
  __shared__ __align__(16) bf16 As[2][BM][LP];
  __shared__ __align__(16) bf16 Bs[2][BN][LP];

  const int bi = blockIdx.z;
  const int m0 = blockIdx.y * BM;
  const int n0 = blockIdx.x * BN;
  const int tid = threadIdx.x;
  const int wave = tid >> 5;
  const int lane = tid & 31;
  const int half = lane >> 4;
  const int idx  = lane & 15;

  auto stage = [&](int bufi, int k0) {
#pragma unroll
    for (int i = 0; i < 4; ++i) {  // A tile: 1024 b128 vectors
      const int v  = tid + i * 256;
      const int r  = v >> 3;
      const int c8 = v & 7;
      size_t row;
      if (MODE == 0)      row = (size_t)bi * TDIM + m0 + r;
      else if (MODE == 1) row = (size_t)(((m0 + r) >> 6) * 192 + 128 + ((m0 + r) & 63));
      else                row = (size_t)(m0 + r);
      cp_b128(&Ag[row * K + k0 + c8 * 8], &As[bufi][r][c8 * 8]);
    }
#pragma unroll
    for (int i = 0; i < 2; ++i) {  // B tile: 512 vectors
      const int v  = tid + i * 256;
      const int r  = v >> 3;
      const int c8 = v & 7;
      size_t row;
      if (MODE == 0) row = (size_t)(((n0 + r) >> 7) * 192 + ((n0 + r) & 127));
      else           row = (size_t)bi * TDIM + n0 + r;
      cp_b128(&Bg[row * K + k0 + c8 * 8], &Bs[bufi][r][c8 * 8]);
    }
  };

  v8f acc[4];
#pragma unroll
  for (int j = 0; j < 4; ++j)
#pragma unroll
    for (int e = 0; e < 8; ++e) acc[j][e] = 0.f;

  stage(0, 0);
  for (int kt = 0; kt < NK; ++kt) {
    wait_async0();
    __syncthreads();
    if (kt + 1 < NK) stage((kt + 1) & 1, (kt + 1) * BK);
    const bf16(*Ac)[LP] = As[kt & 1];
    const bf16(*Bc)[LP] = Bs[kt & 1];
#pragma unroll
    for (int kk = 0; kk < BK; kk += 32) {
      const v16bf a = ld_frag_k(&Ac[wave * 16][kk], LP);
#pragma unroll
      for (int j = 0; j < 4; ++j) {
        const v16bf b = ld_frag_k(&Bc[j * 16][kk], LP);
        acc[j] = wmma_bf16(a, b, acc[j]);
      }
    }
  }

  // epilogue: D element (m = m0 + wave*16 + v + 8*half, n = n0 + j*16 + idx)
#pragma unroll
  for (int j = 0; j < 4; ++j) {
    const int n = n0 + j * 16 + idx;
#pragma unroll
    for (int v = 0; v < 8; ++v) {
      const int m = m0 + wave * 16 + v + 8 * half;
      float val = acc[j][v];
      if (MODE == 0) {
        val += bias[(n >> 7) * 192 + (n & 127)];
        ((bf16*)outp)[((size_t)bi * TDIM + m) * (2 * CDIM) + n] = (bf16)val;
      } else if (MODE == 1) {
        val += bias[(m >> 6) * 192 + 128 + (m & 63)];
        ((bf16*)outp)[((size_t)bi * CDIM + m) * TDIM + n] = (bf16)val;
      } else {
        const size_t o = ((size_t)bi * CDIM + m) * TDIM + n;
        ((float*)outp)[o] = val + bias[m] + resid[o];
      }
    }
  }
}

// ---------------- Flash attention: grid (T/128, B*HEADS) ----------------
// qkT[b][t][h*128 +0..63]=q, +64..127]=k; v[b][c][t]; out rT[b][t][c].
// Q fragments live in registers for the whole block; K/V double-buffered async.
__global__ __launch_bounds__(256) void attn_kernel(const bf16* __restrict__ qkT,
                                                   const bf16* __restrict__ vbuf,
                                                   bf16* __restrict__ rT) {
  constexpr int LP = CH + 8;  // 72-elem pitch
  __shared__ __align__(16) bf16 Ks[2][64][LP];   // rows s, cols c
  __shared__ __align__(16) bf16 Vs[2][64][LP];   // rows c, cols s
  __shared__ __align__(16) bf16 Ps[8][16][LP];

  const int bh = blockIdx.y;
  const int bi = bh >> 3;
  const int hh = bh & 7;
  const int t0 = blockIdx.x * 128;
  const int tid = threadIdx.x;
  const int wave = tid >> 5;
  const int lane = tid & 31;
  const int half = lane >> 4;
  const int idx  = lane & 15;

  const bf16* qbase = qkT + (size_t)bi * TDIM * (2 * CDIM) + hh * 128;
  const bf16* kbase = qbase + 64;
  const bf16* vbase = vbuf + ((size_t)bi * CDIM + hh * CH) * TDIM;

  auto stage_kv = [&](int bufi, int s0) {
#pragma unroll
    for (int i = 0; i < 2; ++i) {
      const int v  = tid + i * 256;
      const int r  = v >> 3;
      const int c8 = v & 7;
      cp_b128(&kbase[(size_t)(s0 + r) * (2 * CDIM) + c8 * 8], &Ks[bufi][r][c8 * 8]);
      cp_b128(&vbase[(size_t)r * TDIM + s0 + c8 * 8], &Vs[bufi][r][c8 * 8]);
    }
  };

  stage_kv(0, 0);

  // Q fragments in registers: A layout, lane idx = row t, two K-halves of CH=64
  v16bf qa[2];
  {
    const bf16* qrow = qbase + (size_t)(t0 + wave * 16 + idx) * (2 * CDIM) + 8 * half;
#pragma unroll
    for (int kf = 0; kf < 2; ++kf) {
      union { u32x4 u[2]; v16bf v; } t;
      t.u[0] = *(const u32x4*)(qrow + kf * 32);
      t.u[1] = *(const u32x4*)(qrow + kf * 32 + 16);
      qa[kf] = t.v;
    }
  }

  float mrow[8], lrow[8];
#pragma unroll
  for (int v = 0; v < 8; ++v) { mrow[v] = -3.0e38f; lrow[v] = 0.f; }
  v8f O[4];
#pragma unroll
  for (int j = 0; j < 4; ++j)
#pragma unroll
    for (int e = 0; e < 8; ++e) O[j][e] = 0.f;

  for (int st = 0; st < TDIM / 64; ++st) {
    wait_async0();
    __syncthreads();
    if (st + 1 < TDIM / 64) stage_kv((st + 1) & 1, (st + 1) * 64);
    const bf16(*Kc)[LP] = Ks[st & 1];
    const bf16(*Vc)[LP] = Vs[st & 1];

    // S[t,s] = q . k  (A from registers; B: n=s, k=c from Kc)
    v8f S[4];
#pragma unroll
    for (int j = 0; j < 4; ++j)
#pragma unroll
      for (int e = 0; e < 8; ++e) S[j][e] = 0.f;
#pragma unroll
    for (int kf = 0; kf < 2; ++kf) {
#pragma unroll
      for (int j = 0; j < 4; ++j) {
        const v16bf b = ld_frag_k(&Kc[j * 16][kf * 32], LP);
        S[j] = wmma_bf16(qa[kf], b, S[j]);
      }
    }
#pragma unroll
    for (int j = 0; j < 4; ++j)
#pragma unroll
      for (int e = 0; e < 8; ++e) S[j][e] *= 0.125f;  // (ch^-1/4)^2

    // online softmax; a row spans the 16 lanes of each half
#pragma unroll
    for (int v = 0; v < 8; ++v) {
      float mx = S[0][v];
#pragma unroll
      for (int j = 1; j < 4; ++j) mx = fmaxf(mx, S[j][v]);
#pragma unroll
      for (int msk = 1; msk < 16; msk <<= 1) mx = fmaxf(mx, __shfl_xor(mx, msk, 32));
      const float newm = fmaxf(mrow[v], mx);
      const float psc  = __expf(mrow[v] - newm);
      mrow[v] = newm;
      float rsum = 0.f;
      const int prow = v + 8 * half;
#pragma unroll
      for (int j = 0; j < 4; ++j) {
        const float p = __expf(S[j][v] - newm);
        rsum += p;
        Ps[wave][prow][j * 16 + idx] = (bf16)p;  // wave-local, DS in-order
      }
#pragma unroll
      for (int msk = 1; msk < 16; msk <<= 1) rsum += __shfl_xor(rsum, msk, 32);
      lrow[v] = lrow[v] * psc + rsum;
#pragma unroll
      for (int j = 0; j < 4; ++j) O[j][v] *= psc;
    }

    // O[t,c] += P . V^T  (A: m=t,k=s from Ps; B: n=c,k=s from Vc)
#pragma unroll
    for (int k0 = 0; k0 < 64; k0 += 32) {
      const v16bf a = ld_frag_k(&Ps[wave][0][k0], LP);
#pragma unroll
      for (int j = 0; j < 4; ++j) {
        const v16bf b = ld_frag_k(&Vc[j * 16][k0], LP);
        O[j] = wmma_bf16(a, b, O[j]);
      }
    }
  }

  // normalize + store rT[b][t][h*64 + c] (lanes vary c -> coalesced)
#pragma unroll
  for (int v = 0; v < 8; ++v) {
    const float inv = 1.0f / lrow[v];
    const int t = t0 + wave * 16 + v + 8 * half;
#pragma unroll
    for (int j = 0; j < 4; ++j) {
      const int cc = j * 16 + idx;
      rT[((size_t)bi * TDIM + t) * CDIM + hh * CH + cc] = (bf16)(O[j][v] * inv);
    }
  }
}

// ---------------- host launcher ----------------
extern "C" void kernel_launch(void* const* d_in, const int* in_sizes, int n_in,
                              void* d_out, int out_size, void* d_ws, size_t ws_size,
                              hipStream_t stream) {
  const float* x        = (const float*)d_in[0];
  const float* gn_scale = (const float*)d_in[1];
  const float* gn_bias  = (const float*)d_in[2];
  const float* w_qkv    = (const float*)d_in[3];
  const float* b_qkv    = (const float*)d_in[4];
  const float* w_proj   = (const float*)d_in[5];
  const float* b_proj   = (const float*)d_in[6];
  float* out = (float*)d_out;

  char* ws = (char*)d_ws;
  size_t off = 0;
  bf16* wqkv_bf  = (bf16*)(ws + off); off += (size_t)(3 * CDIM) * CDIM * 2;
  bf16* wproj_bf = (bf16*)(ws + off); off += (size_t)CDIM * CDIM * 2;
  off = (off + 255) & ~(size_t)255;
  bf16* hT  = (bf16*)(ws + off); off += (size_t)BATCH * TDIM * CDIM * 2;
  bf16* qkT = (bf16*)(ws + off); off += (size_t)BATCH * TDIM * (2 * CDIM) * 2;
  bf16* vbf = (bf16*)(ws + off); off += (size_t)BATCH * CDIM * TDIM * 2;
  bf16* rT  = (bf16*)(ws + off); off += (size_t)BATCH * TDIM * CDIM * 2;
  (void)ws_size; (void)n_in; (void)in_sizes; (void)out_size;

  { // weights -> bf16
    int n1 = 3 * CDIM * CDIM;
    f2bf_kernel<<<(n1 + 255) / 256, 256, 0, stream>>>(w_qkv, wqkv_bf, n1);
    int n2 = CDIM * CDIM;
    f2bf_kernel<<<(n2 + 255) / 256, 256, 0, stream>>>(w_proj, wproj_bf, n2);
  }
  groupnorm_kernel<<<BATCH * NGROUPS, 256, 0, stream>>>(x, gn_scale, gn_bias, hT);
  {
    dim3 grid((2 * CDIM) / 64, TDIM / 128, BATCH);
    gemm_kernel<0><<<grid, 256, 0, stream>>>(hT, wqkv_bf, b_qkv, nullptr, (void*)qkT);
  }
  {
    dim3 grid(TDIM / 64, CDIM / 128, BATCH);
    gemm_kernel<1><<<grid, 256, 0, stream>>>(wqkv_bf, hT, b_qkv, nullptr, (void*)vbf);
  }
  {
    dim3 grid(TDIM / 128, BATCH * NHEADS);
    attn_kernel<<<grid, 256, 0, stream>>>(qkT, vbf, rT);
  }
  {
    dim3 grid(TDIM / 64, CDIM / 128, BATCH);
    gemm_kernel<2><<<grid, 256, 0, stream>>>(wproj_bf, rT, b_proj, x, (void*)out);
  }
}
